// SwitchableLSTMModelV3_85693187490432
// MI455X (gfx1250) — matrix operand order repaired and checked by reading, not compiled, not run on hardware
//
#include <hip/hip_runtime.h>

typedef __attribute__((ext_vector_type(8)))  _Float16 v8h;
typedef __attribute__((ext_vector_type(16))) _Float16 v16h;
typedef __attribute__((ext_vector_type(8)))  float    v8f;

#define HID   500
#define NP    512          // padded per-gate width
#define KP    1024         // padded K: [h(512) | x(512)]
#define BATCH 128
#define SEQT  400
#define LROW  520          // LDS B-row pitch in halves (512 + 8 pad -> bank spread)
#define LHALF (64 * LROW)  // halves per LDS buffer
#define LDSB  (2 * LHALF * 2)

__device__ __forceinline__ float sigmoidf_(float x) {
  return 1.0f / (1.0f + __expf(-x));
}

// Build Wcat[2048][1024] f16: row r = g*512+n -> [Whh[g*500+n, 0:500] 0pad | Wih[g*500+n, 0:Din] 0pad]
// bias[2048] = bih+bhh (0 on pad rows). Gate order (PyTorch): i,f,g,o.
__global__ void __launch_bounds__(256)
prep_weights(const float* __restrict__ Wih, const float* __restrict__ Whh,
             const float* __restrict__ bih, const float* __restrict__ bhh,
             _Float16* __restrict__ Wcat, float* __restrict__ bias, int Din)
{
  size_t idx = (size_t)blockIdx.x * 256 + threadIdx.x;   // over 2048*1024 exactly
  int r = (int)(idx >> 10);
  int k = (int)(idx & (KP - 1));
  int g = r >> 9;
  int n = r & (NP - 1);
  _Float16 v = (_Float16)0.0f;
  if (n < HID) {
    int wr = g * HID + n;
    if (k < HID)                      v = (_Float16)Whh[(size_t)wr * HID + k];
    else if (k >= NP && k < NP + Din) v = (_Float16)Wih[(size_t)wr * Din + (k - NP)];
  }
  Wcat[idx] = v;
  if (k == 0) bias[r] = (n < HID) ? (bih[g * HID + n] + bhh[g * HID + n]) : 0.0f;
}

// Per-layer init: A0 = [h0=0 | x_0 | 0pad], A1 = 0, c = 0.
__global__ void __launch_bounds__(256)
layer_init(_Float16* __restrict__ A0, _Float16* __restrict__ A1, float* __restrict__ Cst,
           const float* __restrict__ Xin32, const _Float16* __restrict__ Xin16, int Din)
{
  int idx = blockIdx.x * 256 + threadIdx.x;              // over 128*1024 exactly
  int b = idx >> 10;
  int k = idx & (KP - 1);
  _Float16 v = (_Float16)0.0f;
  int col = k - NP;
  if (col >= 0 && col < Din) {
    v = Xin32 ? (_Float16)Xin32[(size_t)b * SEQT * Din + col]
              : Xin16[(size_t)b * SEQT * Din + col];
  }
  A0[idx] = v;
  A1[idx] = (_Float16)0.0f;
  if (k < NP) Cst[b * NP + k] = 0.0f;
}

// One timestep: gates[128,2048] = A[128,1024] @ Wcat^T, then LSTM cell.
// Block nc owns hidden units [16*nc, 16*nc+16) x 4 gates (64 Wcat rows, async-staged in LDS,
// double-buffered across the two 512-wide K chunks). Wave w owns batch rows [16w,16w+16).
__global__ void __launch_bounds__(256)
lstm_step(const _Float16* __restrict__ Wcat, const float* __restrict__ bias,
          const _Float16* __restrict__ A, _Float16* __restrict__ Anext,
          float* __restrict__ Cst,
          const float* __restrict__ Xin32, const _Float16* __restrict__ Xin16,
          _Float16* __restrict__ Xout, int t, int Din, int last)
{
  extern __shared__ __align__(16) _Float16 lbs[];        // [2][64][LROW]
  const int tid  = threadIdx.x;
  const int nc   = blockIdx.x;                           // 0..31
  const int wv   = tid >> 5;                             // wave = M-tile (0..7)
  const int ln   = tid & 31;
  const int lrow = ln & 15;                              // tile row (A) / tile col (B,C,D)
  const int lhi  = ln >> 4;

  // Issue async Global->LDS staging of one 512-wide K chunk into LDS buffer `buf`.
  // 64 rows x 512 halves = 4096 16B transfers; 16 per thread (uniform, full EXEC).
  auto stage_async = [&](int kc, int buf) {
    for (int idx = tid; idx < 64 * 64; idx += 256) {
      int r  = idx >> 6;
      int ch = idx & 63;
      int g  = r >> 4, i = r & 15;
      const void* src =
          (const void*)(Wcat + ((size_t)(g * NP + nc * 16 + i)) * KP + kc + ch * 8);
      unsigned dst =
          (unsigned)(uintptr_t)(void*)(lbs + (size_t)buf * LHALF + r * LROW + ch * 8);
      asm volatile("global_load_async_to_lds_b128 %0, %1, off"
                   :: "v"(dst), "v"(src) : "memory");
    }
  };

  v8f acc0 = {}, acc1 = {}, acc2 = {}, acc3 = {};
  const _Float16* Arow = A + (size_t)(wv * 16 + lrow) * KP;

  stage_async(0, 0);
  asm volatile("s_wait_asynccnt 0x0" ::: "memory");
  __syncthreads();

  for (int half = 0; half < 2; ++half) {
    if (half == 0) stage_async(512, 1);                  // prefetch chunk 1 while computing chunk 0
    const _Float16* lb = lbs + (size_t)half * LHALF;
    const int kc = half * 512;

    for (int kk2 = 0; kk2 < 512; kk2 += 32) {
      // --- A fragment (contiguous K, matching B): lane<16 -> K 0..15, lane>=16 -> K 16..31
      int ka = kc + kk2 + 16 * lhi;
      v8h a_lo = *(const v8h*)(Arow + ka);
      v8h a_hi = *(const v8h*)(Arow + ka + 8);
      v16h a = __builtin_shufflevector(a_lo, a_hi, 0, 1, 2, 3, 4, 5, 6, 7, 8, 9, 10, 11, 12, 13, 14, 15);

      // --- B fragments from LDS: lane holds column n=lrow, contiguous K (lane<16: K 0..15, lane>=16: 16..31)
      int kb = kk2 + 16 * lhi;
      const _Float16* b0p = lb + (0 * 16 + lrow) * LROW + kb;
      const _Float16* b1p = lb + (1 * 16 + lrow) * LROW + kb;
      const _Float16* b2p = lb + (2 * 16 + lrow) * LROW + kb;
      const _Float16* b3p = lb + (3 * 16 + lrow) * LROW + kb;
      v16h b0 = __builtin_shufflevector(*(const v8h*)b0p, *(const v8h*)(b0p + 8),
                                        0, 1, 2, 3, 4, 5, 6, 7, 8, 9, 10, 11, 12, 13, 14, 15);
      v16h b1 = __builtin_shufflevector(*(const v8h*)b1p, *(const v8h*)(b1p + 8),
                                        0, 1, 2, 3, 4, 5, 6, 7, 8, 9, 10, 11, 12, 13, 14, 15);
      v16h b2 = __builtin_shufflevector(*(const v8h*)b2p, *(const v8h*)(b2p + 8),
                                        0, 1, 2, 3, 4, 5, 6, 7, 8, 9, 10, 11, 12, 13, 14, 15);
      v16h b3 = __builtin_shufflevector(*(const v8h*)b3p, *(const v8h*)(b3p + 8),
                                        0, 1, 2, 3, 4, 5, 6, 7, 8, 9, 10, 11, 12, 13, 14, 15);

      acc0 = __builtin_amdgcn_wmma_f32_16x16x32_f16(false, a, false, b0, (short)0, acc0, false, false);
      acc1 = __builtin_amdgcn_wmma_f32_16x16x32_f16(false, a, false, b1, (short)0, acc1, false, false);
      acc2 = __builtin_amdgcn_wmma_f32_16x16x32_f16(false, a, false, b2, (short)0, acc2, false, false);
      acc3 = __builtin_amdgcn_wmma_f32_16x16x32_f16(false, a, false, b3, (short)0, acc3, false, false);
    }

    if (half == 0) {
      asm volatile("s_wait_asynccnt 0x0" ::: "memory");
      __syncthreads();
    }
  }

  // --- LSTM cell (wave-local; C/D layout: lane -> col n, VGPR j -> row 8*lhi + j)
  int n = nc * 16 + lrow;
  if (n < HID) {
    float bi = bias[n], bf = bias[NP + n], bg = bias[2 * NP + n], bo = bias[3 * NP + n];
#pragma unroll
    for (int j = 0; j < 8; ++j) {
      int b = wv * 16 + 8 * lhi + j;
      float iv = sigmoidf_(acc0[j] + bi);
      float fv = sigmoidf_(acc1[j] + bf);
      float gv = tanhf(acc2[j] + bg);
      float ov = sigmoidf_(acc3[j] + bo);
      float c  = fv * Cst[b * NP + n] + iv * gv;
      Cst[b * NP + n] = c;
      float h  = ov * tanhf(c);
      _Float16 hh = (_Float16)h;
      Xout[((size_t)b * SEQT + t) * HID + n] = hh;
      if (!last) Anext[(size_t)b * KP + n] = hh;
    }
  }

  // --- prepare x-region of Anext for step t+1 (this block's 16 columns)
  if (!last) {
    int c0 = nc * 16;
    for (int idx = tid; idx < 16 * BATCH; idx += 256) {
      int col = c0 + (idx & 15);
      int b   = idx >> 4;
      if (col < Din) {
        _Float16 v = Xin32 ? (_Float16)Xin32[((size_t)b * SEQT + (t + 1)) * Din + col]
                           : Xin16[((size_t)b * SEQT + (t + 1)) * Din + col];
        Anext[(size_t)b * KP + NP + col] = v;
      }
    }
  }
}

// out[128,2] = h_last @ fcW^T + fcb
__global__ void __launch_bounds__(256)
fc_kernel(const _Float16* __restrict__ X, const float* __restrict__ fcW,
          const float* __restrict__ fcb, float* __restrict__ out)
{
  int tid = threadIdx.x;                                  // 256 = 128*2 outputs
  int b = tid >> 1, o = tid & 1;
  const _Float16* h = X + ((size_t)b * SEQT + (SEQT - 1)) * HID;
  float s = fcb[o];
  for (int k = 0; k < HID; ++k) s += (float)h[k] * fcW[o * HID + k];
  out[b * 2 + o] = s;
}

extern "C" void kernel_launch(void* const* d_in, const int* in_sizes, int n_in,
                              void* d_out, int out_size, void* d_ws, size_t ws_size,
                              hipStream_t stream) {
  (void)in_sizes; (void)n_in; (void)out_size; (void)ws_size;
  const float* x = (const float*)d_in[0];
  const float *Wih[4], *Whh[4], *bih[4], *bhh[4];
  for (int l = 0; l < 4; ++l) {
    Wih[l] = (const float*)d_in[1 + 4 * l];
    Whh[l] = (const float*)d_in[2 + 4 * l];
    bih[l] = (const float*)d_in[3 + 4 * l];
    bhh[l] = (const float*)d_in[4 + 4 * l];
  }
  const float* fcW = (const float*)d_in[17];
  const float* fcb = (const float*)d_in[18];

  char* ws = (char*)d_ws;
  size_t off = 0;
  _Float16* Wcat[4]; float* bias[4];
  for (int l = 0; l < 4; ++l) { Wcat[l] = (_Float16*)(ws + off); off += (size_t)2048 * KP * 2; }
  for (int l = 0; l < 4; ++l) { bias[l] = (float*)(ws + off);    off += 2048 * 4; }
  _Float16* A0 = (_Float16*)(ws + off); off += (size_t)BATCH * KP * 2;
  _Float16* A1 = (_Float16*)(ws + off); off += (size_t)BATCH * KP * 2;
  float*    Cs = (float*)(ws + off);    off += (size_t)BATCH * NP * 4;
  _Float16* Xb[2];
  Xb[0] = (_Float16*)(ws + off); off += (size_t)BATCH * SEQT * HID * 2;
  Xb[1] = (_Float16*)(ws + off); off += (size_t)BATCH * SEQT * HID * 2;

  for (int l = 0; l < 4; ++l) {
    int Din = (l == 0) ? 11 : HID;
    prep_weights<<<8192, 256, 0, stream>>>(Wih[l], Whh[l], bih[l], bhh[l], Wcat[l], bias[l], Din);
  }

  for (int l = 0; l < 4; ++l) {
    int Din = (l == 0) ? 11 : HID;
    const float*    xin32 = (l == 0) ? x : nullptr;
    const _Float16* xin16 = (l == 0) ? nullptr : Xb[(l - 1) & 1];
    _Float16*       xout  = Xb[l & 1];
    layer_init<<<512, 256, 0, stream>>>(A0, A1, Cs, xin32, xin16, Din);
    for (int t = 0; t < SEQT; ++t) {
      const _Float16* Acur = (t & 1) ? A1 : A0;
      _Float16*       Anxt = (t & 1) ? A0 : A1;
      lstm_step<<<32, 256, LDSB, stream>>>(Wcat[l], bias[l], Acur, Anxt, Cs,
                                           xin32, xin16, xout, t, Din, t == SEQT - 1);
    }
  }

  fc_kernel<<<1, 256, 0, stream>>>(Xb[1], fcW, fcb, (float*)d_out);
}
